// StreamableLSTM_82085414961673
// MI455X (gfx1250) — matrix operand order repaired and checked by reading, not compiled, not run on hardware
//
#include <hip/hip_runtime.h>

// ---------------------------------------------------------------------------
// Stacked LSTM (L=2, B=16, T=2048, D=512) for gfx1250 / MI455X.
//  Phase A (parallel):  xg = y @ Wx + b   -> WMMA bf16 GEMM, TDM-staged A.
//  Phase B (sequential): per-step h @ Wh with Wh B-operands pinned in VGPRs,
//     8 co-resident workgroups, grid barrier via monotonic atomic counter,
//     h exchanged through L2 and pulled into LDS each step by the
//     Tensor Data Mover (tensor_load_to_lds + s_wait_tensorcnt).
// ---------------------------------------------------------------------------

typedef __attribute__((ext_vector_type(16))) __bf16 v16bf;
typedef __attribute__((ext_vector_type(8)))  float  v8f;
typedef __attribute__((ext_vector_type(4)))  unsigned int v4u;
typedef __attribute__((ext_vector_type(8)))  int  v8i;
typedef __attribute__((ext_vector_type(4)))  int  v4i;

#define Bsz 16
#define Tsz 2048
#define Dsz 512
#define G4D 2048          // 4*D
#define NT4 128           // G4D/16 gate tiles
#define KC  16            // D/32 k-chunks
#define NWG_REC 8
#define THR_REC 512
#define ROWP 520          // padded LDS row length in ushorts (1040B -> bank-spread)

__device__ __forceinline__ unsigned short f2bf(float f) {
  union { float f; unsigned u; } v; v.f = f;
  unsigned u = v.u;
  u += 0x7fffu + ((u >> 16) & 1u);          // round-to-nearest-even
  return (unsigned short)(u >> 16);
}
__device__ __forceinline__ float bf2f(unsigned short h) {
  union { unsigned u; float f; } v; v.u = ((unsigned)h) << 16;
  return v.f;
}
__device__ __forceinline__ v16bf mk16(uint4 a, uint4 b) {
  union { uint4 q[2]; v16bf v; } u; u.q[0] = a; u.q[1] = b; return u.v;
}
__device__ __forceinline__ v8f vzero8() {
  v8f v;
#pragma unroll
  for (int i = 0; i < 8; ++i) v[i] = 0.0f;
  return v;
}
__device__ __forceinline__ uint4 packbf8(const float* s) {
  uint4 q;
  q.x = (unsigned)f2bf(s[0]) | ((unsigned)f2bf(s[1]) << 16);
  q.y = (unsigned)f2bf(s[2]) | ((unsigned)f2bf(s[3]) << 16);
  q.z = (unsigned)f2bf(s[4]) | ((unsigned)f2bf(s[5]) << 16);
  q.w = (unsigned)f2bf(s[6]) | ((unsigned)f2bf(s[7]) << 16);
  return q;
}
__device__ __forceinline__ float sigm(float x)  { return 1.0f / (1.0f + __expf(-x)); }
__device__ __forceinline__ float tanhfa(float x){ return 2.0f / (1.0f + __expf(-2.0f * x)) - 1.0f; }

// Raw LDS byte offset of a __shared__ object (addrspace(3) pointers are the
// 32-bit offset from the workgroup's LDS base on amdgcn).
__device__ __forceinline__ unsigned lds_off(const void* p) {
  return (unsigned)(unsigned long long)
      (__attribute__((address_space(3))) const void*)p;
}

// Issue one TDM load of a 2-D tile: 16 rows x 512 bf16 elements, row stride
// 512 elements in memory, destination LDS padded by 16B per 1024B row
// (pad_interval=256 dwords -> code 7, pad_amount=4 dwords -> code 3).
// Whole-wave instruction: call from exactly one wave per workgroup.
__device__ __forceinline__ void tdm_load_16x512(unsigned ldsByte, const void* gptr) {
  unsigned long long ga = (unsigned long long)gptr;
  v4u g0;
  g0[0] = 1u;                                          // count=1, user mode
  g0[1] = ldsByte;                                     // lds_addr (bytes)
  g0[2] = (unsigned)(ga & 0xffffffffu);                // global_addr[31:0]
  g0[3] = (unsigned)((ga >> 32) & 0x01ffffffu)         // global_addr[56:32]
        | 0x80000000u;                                 // type=2 ("image")
  v8i g1;
  g1[0] = (int)(0x00010000u                            // data_size=1 (2B)
        | (1u << 20)                                   // pad_enable
        | (7u << 22)                                   // pad_interval: 256 dw
        | (3u << 25));                                 // pad_amount: 4 dw
  g1[1] = (int)(512u << 16);                           // tensor_dim0 = 512
  g1[2] = (int)(16u << 16);                            // tensor_dim1 = 16
  g1[3] = (int)(512u << 16);                           // tile_dim0 = 512
  g1[4] = 16;                                          // tile_dim1=16, dim2=0
  g1[5] = 512;                                         // tensor_dim0_stride
  g1[6] = 0;
  g1[7] = 0;
  v4i gz;
  gz[0] = 0; gz[1] = 0; gz[2] = 0; gz[3] = 0;
  v8i gz8;
#pragma unroll
  for (int i = 0; i < 8; ++i) gz8[i] = 0;
  __builtin_amdgcn_tensor_load_to_lds(g0, g1, gz, gz, gz8,
                                      16 /*cpol: scope=DEV*/);
}

// ---- x[B,T,D] f32 -> y0[T,B,D] bf16; also zero the grid-barrier counter ----
__global__ void k_permute(const float* __restrict__ x, unsigned short* __restrict__ y0,
                          int* __restrict__ cnt) {
  if (blockIdx.x == 0 && threadIdx.x == 0)
    __hip_atomic_store(cnt, 0, __ATOMIC_RELAXED, __HIP_MEMORY_SCOPE_AGENT);
  size_t base = ((size_t)blockIdx.x * blockDim.x + threadIdx.x) * 8;
  int b = (int)(base >> 20);            // / (T*D) = 2^20
  int t = (int)(base >> 9) & 2047;
  int d = (int)base & 511;
  const float4* xp = (const float4*)(x + base);
  float4 f0 = xp[0], f1 = xp[1];
  float s[8] = {f0.x, f0.y, f0.z, f0.w, f1.x, f1.y, f1.z, f1.w};
  *(uint4*)(y0 + ((size_t)(t * 16 + b)) * 512 + d) = packbf8(s);
}

// ---- Pack Wx/Wh (f32 [L,D,4D]) into WMMA B-operand lane order, bf16 --------
// block(nt,kc): lane l -> column N = nt*16 + (l&15), K = kc*32 + (l>>4)*8 + (e&7) + (e>>3)*16
__global__ void k_packw(const float* __restrict__ Wx, const float* __restrict__ Wh,
                        unsigned short* __restrict__ WxP, unsigned short* __restrict__ WhP) {
  unsigned id = blockIdx.x * blockDim.x + threadIdx.x;
  int lane  = id & 31;
  int kc    = (id >> 5) & 15;
  int nt    = (id >> 9) & 127;
  int layer = (id >> 16) & 1;
  int which = id >> 17;
  const float* W = (which ? Wh : Wx) + (size_t)layer * Dsz * G4D;
  unsigned short* P = (which ? WhP : WxP) + (size_t)layer * (NT4 * KC * 512);
  int n  = nt * 16 + (lane & 15);
  int kb = kc * 32 + (lane >> 4) * 8;
  __align__(16) unsigned short tmp[16];
#pragma unroll
  for (int e = 0; e < 16; ++e) tmp[e] = f2bf(W[(size_t)(kb + (e & 7) + (e >> 3) * 16) * G4D + n]);
  uint4* dst = (uint4*)(P + (size_t)(nt * 16 + kc) * 512 + lane * 16);
  dst[0] = *(uint4*)&tmp[0];
  dst[1] = *(uint4*)&tmp[8];
}

// ---- Phase A: xg[t][nt][lane][8] = (y @ Wx + b) in WMMA C-tile order -------
__global__ void __launch_bounds__(256) k_gemm_xg(const unsigned short* __restrict__ y,
                                                 const unsigned short* __restrict__ WxPl,
                                                 const float* __restrict__ biasl,
                                                 unsigned short* __restrict__ xg) {
  __shared__ __align__(16) unsigned short aS[16 * ROWP];
  int t = blockIdx.x;                       // one M-tile == one timestep (B=16)
  int tid = threadIdx.x;
  if (tid < 32) {                           // wave 0: TDM-stage the A tile
    tdm_load_16x512(lds_off(aS), y + (size_t)t * 16 * 512);
    __builtin_amdgcn_s_wait_tensorcnt(0);
  }
  __syncthreads();
  int wv = tid >> 5, lane = tid & 31;
  int nt = blockIdx.y * 8 + wv;
  v8f acc = vzero8();
  int rowbase = (lane & 15) * ROWP;
#pragma unroll
  for (int kc = 0; kc < KC; ++kc) {
    int k0 = kc * 32 + (lane >> 4) * 8;                 // 16-bit A-layout
    uint4 a0 = *(const uint4*)&aS[rowbase + k0];
    uint4 a1 = *(const uint4*)&aS[rowbase + k0 + 16];
    const uint4* bp = (const uint4*)(WxPl + (size_t)(nt * 16 + kc) * 512 + lane * 16);
    acc = __builtin_amdgcn_wmma_f32_16x16x32_bf16(false, mk16(a0, a1),
                                                  false, mk16(bp[0], bp[1]),
                                                  (short)0, acc, false, false);
  }
  float bv = biasl[nt * 16 + (lane & 15)];
  float s[8];
#pragma unroll
  for (int r = 0; r < 8; ++r) s[r] = acc[r] + bv;
  *(uint4*)(xg + (((size_t)t * NT4 + nt) * 32 + lane) * 8) = packbf8(s);
}

// ---- Phase B: recurrence. 8 WGs x 16 waves; wave -> one gate tile ----------
__global__ void __launch_bounds__(THR_REC) k_recur(const unsigned short* __restrict__ xg,
                                                   const unsigned short* __restrict__ WhPl,
                                                   unsigned short* __restrict__ hG,
                                                   unsigned short* __restrict__ yNext,
                                                   int* __restrict__ cnt, int barBase) {
  __shared__ __align__(16) unsigned short hS[16 * ROWP];
  __shared__ float gbuf[16][32][8];
  int tid  = threadIdx.x;
  int wv   = tid >> 5, lane = tid & 31;
  int ht   = blockIdx.x * 4 + (wv >> 2);   // hidden tile 0..31
  int gate = wv & 3;                       // i,f,g,o
  int nt   = gate * 32 + ht;               // gate-tile column in 4D

  // Pin this wave's Wh B-operands in VGPRs for the whole sequence (16 KB/wave)
  v16bf wb[KC];
#pragma unroll
  for (int kc = 0; kc < KC; ++kc) {
    const uint4* bp = (const uint4*)(WhPl + (size_t)(nt * 16 + kc) * 512 + lane * 16);
    wb[kc] = mk16(bp[0], bp[1]);
  }
  for (int i = tid; i < 16 * ROWP; i += THR_REC) hS[i] = 0;   // h_{-1} = 0
  float cc[8];
#pragma unroll
  for (int r = 0; r < 8; ++r) cc[r] = 0.0f;                   // c_{-1} = 0
  unsigned hSoff = lds_off(hS);
  __syncthreads();

  int rowbase = (lane & 15) * ROWP;
  for (int t = 0; t < Tsz; ++t) {
    v8f acc = vzero8();
#pragma unroll
    for (int kc = 0; kc < KC; ++kc) {
      int k0 = kc * 32 + (lane >> 4) * 8;
      uint4 a0 = *(const uint4*)&hS[rowbase + k0];
      uint4 a1 = *(const uint4*)&hS[rowbase + k0 + 16];
      acc = __builtin_amdgcn_wmma_f32_16x16x32_bf16(false, mk16(a0, a1),
                                                    false, wb[kc],
                                                    (short)0, acc, false, false);
    }
    uint4 xq = *(const uint4*)(xg + (((size_t)t * NT4 + nt) * 32 + lane) * 8);
    unsigned xv[4] = {xq.x, xq.y, xq.z, xq.w};
#pragma unroll
    for (int r = 0; r < 8; ++r) {
      unsigned u = xv[r >> 1];
      float xf = bf2f((unsigned short)((r & 1) ? (u >> 16) : (u & 0xffffu)));
      gbuf[wv][lane][r] = acc[r] + xf;
    }
    __syncthreads();
    if (gate == 0) {                       // one wave per hidden tile updates c,h
      int b0 = 8 * (lane >> 4);
      int d  = ht * 16 + (lane & 15);
#pragma unroll
      for (int r = 0; r < 8; ++r) {
        float iv = sigm(gbuf[wv + 0][lane][r]);
        float fv = sigm(gbuf[wv + 1][lane][r]);
        float gv = tanhfa(gbuf[wv + 2][lane][r]);
        float ov = sigm(gbuf[wv + 3][lane][r]);
        float cn = fv * cc[r] + iv * gv;
        cc[r] = cn;
        unsigned short hb = f2bf(ov * tanhfa(cn));
        int b = b0 + r;
        hG[(size_t)(t & 1) * 8192 + b * 512 + d] = hb;          // double-buffered
        yNext[((size_t)t * 16 + b) * 512 + d]    = hb;          // next layer input
      }
    }
    __threadfence();
    __syncthreads();
    if (tid == 0) {                         // grid barrier: monotonic counter
      __hip_atomic_fetch_add(cnt, 1, __ATOMIC_ACQ_REL, __HIP_MEMORY_SCOPE_AGENT);
      int target = barBase + (t + 1) * NWG_REC;
      while (__hip_atomic_load(cnt, __ATOMIC_ACQUIRE, __HIP_MEMORY_SCOPE_AGENT) < target)
        __builtin_amdgcn_s_sleep(2);
    }
    __syncthreads();
    if (wv == 0) {                          // TDM: pull full h_t into LDS
      tdm_load_16x512(hSoff, hG + (size_t)(t & 1) * 8192);
      __builtin_amdgcn_s_wait_tensorcnt(0);
    }
    __syncthreads();
  }
}

// ---- out[b,t,d] = x[b,t,d] + h_last[t,b,d] ---------------------------------
__global__ void k_final(const float* __restrict__ x, const unsigned short* __restrict__ y2,
                        float* __restrict__ out) {
  size_t base = ((size_t)blockIdx.x * blockDim.x + threadIdx.x) * 8;
  int b = (int)(base >> 20);
  int t = (int)(base >> 9) & 2047;
  int d = (int)base & 511;
  uint4 q = *(const uint4*)(y2 + ((size_t)t * 16 + b) * 512 + d);
  unsigned qs[4] = {q.x, q.y, q.z, q.w};
  const float4* xp = (const float4*)(x + base);
  float4 f0 = xp[0], f1 = xp[1];
  float s[8] = {f0.x, f0.y, f0.z, f0.w, f1.x, f1.y, f1.z, f1.w};
  float r[8];
#pragma unroll
  for (int i = 0; i < 8; ++i) {
    unsigned u = qs[i >> 1];
    r[i] = s[i] + bf2f((unsigned short)((i & 1) ? (u >> 16) : (u & 0xffffu)));
  }
  float4* op = (float4*)(out + base);
  op[0] = make_float4(r[0], r[1], r[2], r[3]);
  op[1] = make_float4(r[4], r[5], r[6], r[7]);
}

extern "C" void kernel_launch(void* const* d_in, const int* in_sizes, int n_in,
                              void* d_out, int out_size, void* d_ws, size_t ws_size,
                              hipStream_t stream) {
  const float* x    = (const float*)d_in[0];
  const float* Wx   = (const float*)d_in[1];
  const float* Wh   = (const float*)d_in[2];
  const float* bias = (const float*)d_in[3];
  float* out = (float*)d_out;
  char* ws = (char*)d_ws;

  size_t o = 0;
  int* cnt = (int*)(ws + o);                 o += 256;
  unsigned short* y0  = (unsigned short*)(ws + o); o += (size_t)Tsz * Bsz * Dsz * 2;
  unsigned short* y1  = (unsigned short*)(ws + o); o += (size_t)Tsz * Bsz * Dsz * 2;
  unsigned short* y2  = (unsigned short*)(ws + o); o += (size_t)Tsz * Bsz * Dsz * 2;
  unsigned short* xg  = (unsigned short*)(ws + o); o += (size_t)Tsz * G4D * Bsz * 2;
  unsigned short* WxP = (unsigned short*)(ws + o); o += 2 * (size_t)Dsz * G4D * 2;
  unsigned short* WhP = (unsigned short*)(ws + o); o += 2 * (size_t)Dsz * G4D * 2;
  unsigned short* hG  = (unsigned short*)(ws + o); o += 2 * 8192 * 2;
  (void)o; (void)ws_size; (void)in_sizes; (void)n_in; (void)out_size;

  k_permute<<<8192, 256, 0, stream>>>(x, y0, cnt);
  k_packw<<<1024, 256, 0, stream>>>(Wx, Wh, WxP, WhP);

  unsigned short* yin = y0;
  for (int l = 0; l < 2; ++l) {
    unsigned short* yout = (l == 0) ? y1 : y2;
    k_gemm_xg<<<dim3(Tsz, 16), 256, 0, stream>>>(yin, WxP + (size_t)l * Dsz * G4D,
                                                 bias + l * G4D, xg);
    k_recur<<<NWG_REC, THR_REC, 0, stream>>>(xg, WhP + (size_t)l * Dsz * G4D,
                                             hG, yout, cnt, l * Tsz * NWG_REC);
    yin = yout;
  }
  k_final<<<8192, 256, 0, stream>>>(x, y2, out);
}
